// Disentangler_39737037423086
// MI455X (gfx1250) — matrix operand order repaired and checked by reading, not compiled, not run on hardware
//
#include <hip/hip_runtime.h>
#include <hip/hip_bf16.h>
#include <math.h>

typedef __attribute__((ext_vector_type(16))) _Float16 v16h;
typedef __attribute__((ext_vector_type(8)))  float    v8f;
typedef unsigned int v4u __attribute__((ext_vector_type(4)));
typedef int          v4i __attribute__((ext_vector_type(4)));
typedef int          v8i __attribute__((ext_vector_type(8)));

#define T_  16
#define N_  8192
#define D_  256
#define L_  4
#define C_  64
#define NN_ 4096
#define H_  128          // 2*C
#define TOK_BLK 128      // tokens per block in main kernel
#define CHUNKS  (1024 / TOK_BLK)   // per (p,l,t): NN/L = 1024 tokens

#define W1H_ELEMS (2 * L_ * H_ * D_)        // 262144 halfs
#define HACC_ELEMS (2 * L_ * T_ * H_)       // 16384 floats

#if defined(__gfx1250__) && __has_builtin(__builtin_amdgcn_tensor_load_to_lds) && \
    __has_builtin(__builtin_amdgcn_s_wait_tensorcnt)
#define USE_TDM 1
#else
#define USE_TDM 0
#endif

// ---------------------------------------------------------------------------
// Prep: fp32 W1 -> f16 transposed [pl][n][k]; zero hacc accumulator.
// ---------------------------------------------------------------------------
__global__ __launch_bounds__(256)
void prep_kernel(const float* __restrict__ nW1, const float* __restrict__ eW1,
                 _Float16* __restrict__ w1h, float* __restrict__ hacc) {
  int idx = blockIdx.x * 256 + threadIdx.x;
  if (idx < W1H_ELEMS) {
    int k  = idx & (D_ - 1);
    int n  = (idx >> 8) & (H_ - 1);
    int pl = idx >> 15;                 // p*4 + l
    int p  = pl >> 2, l = pl & 3;
    const float* W = p ? eW1 : nW1;     // [L][D][H]
    w1h[idx] = (_Float16)W[(l * D_ + k) * H_ + n];
  } else {
    int j = idx - W1H_ELEMS;
    if (j < HACC_ELEMS) hacc[j] = 0.0f;
  }
}

// ---------------------------------------------------------------------------
// Main: gather + LayerNorm -> LDS f16 tile; GEMM [128,256]x[256,128] via WMMA;
// bias + exact GELU; column-sum over tokens -> global atomic accumulate.
// W1 tile staged into LDS by the Tensor Data Mover (one descriptor, wave 0).
// ---------------------------------------------------------------------------
__global__ __launch_bounds__(256)
void mlp_kernel(const float* __restrict__ x,
                const float* __restrict__ lnw, const float* __restrict__ lnb,
                const float* __restrict__ nb1, const float* __restrict__ eb1,
                const int* __restrict__ nflat, const int* __restrict__ eflat,
                const int* __restrict__ nhist, const int* __restrict__ ehist,
                const _Float16* __restrict__ w1h, float* __restrict__ hacc) {
  __shared__ _Float16 As[TOK_BLK * D_];   // 64 KB  (tokens x K, f16)
  __shared__ _Float16 Bs[H_ * D_];        // 64 KB  (n x K, f16, transposed W1)
  __shared__ float    hsum[H_];

  const int b     = blockIdx.x;
  const int chunk = b & (CHUNKS - 1);
  const int t     = (b >> 3) & (T_ - 1);
  const int l     = (b >> 7) & (L_ - 1);
  const int p     = b >> 9;

  const int tid    = threadIdx.x;
  const int lane   = tid & 31;
  const int w      = tid >> 5;            // wave id, 0..7
  const int lane_r = lane & 15;
  const int lane_h = lane >> 4;

  const int* hist = (p ? ehist : nhist) + l * 16384 + t * 1024 + chunk * TOK_BLK;
  const int* flat = p ? eflat : nflat;

  if (tid < H_) hsum[tid] = 0.0f;

  // ---- stage W1 slice (64 KB contiguous) into LDS ----
#if USE_TDM
  if (w == 0) {
    // Tensor DMA descriptor: 1-row 2D tile of 16384 dwords (contiguous 64 KB).
    const unsigned lds_off = (unsigned)(size_t)(void*)&Bs[0];   // flat LDS addr[31:0] = LDS offset
    const unsigned long long ga =
        (unsigned long long)(size_t)(const void*)(w1h + (size_t)(p * L_ + l) * H_ * D_);
    v4u g0;
    g0.x = 1u;                                   // count=1, user descriptor
    g0.y = lds_off;                              // lds_addr
    g0.z = (unsigned)ga;                         // global_addr[31:0]
    g0.w = (unsigned)(ga >> 32) | 0x80000000u;   // global_addr[56:32] | type=2<<30
    v8i g1 = {
      0x20000,              // workgroup_mask=0, data_size=2 (4B), no flags
      0x40000000,           // tensor_dim0[15:0]=16384 in [31:16]
      0x00010000,           // tensor_dim0[31:16]=0, tensor_dim1=1
      0x40000000,           // tile_dim0=16384 in [31:16]
      0x00000001,           // tile_dim1=1, tile_dim2=0
      16384,                // tensor_dim0_stride lo
      0, 0                  // stride hi / tensor_dim1_stride (unused)
    };
    v4i gz = { 0, 0, 0, 0 };
#if __clang_major__ >= 23
    v8i gz8 = { 0, 0, 0, 0, 0, 0, 0, 0 };
    __builtin_amdgcn_tensor_load_to_lds(g0, g1, gz, gz, gz8, 0);
#else
    __builtin_amdgcn_tensor_load_to_lds(g0, g1, gz, gz, 0);
#endif
  }
#else
  {
    const uint4* src = (const uint4*)(w1h + (size_t)(p * L_ + l) * H_ * D_);
    uint4* dst = (uint4*)Bs;
#pragma unroll
    for (int i = 0; i < 16; ++i) dst[tid + i * 256] = src[tid + i * 256];
  }
#endif

  // ---- gather + LayerNorm 16 tokens per wave into As ----
  {
    const int k0 = lane * 8;
    float lw[8], lb[8];
#pragma unroll
    for (int e = 0; e < 8; ++e) { lw[e] = lnw[k0 + e]; lb[e] = lnb[k0 + e]; }

    for (int j = 0; j < 16; ++j) {
      const int m   = w * 16 + j;
      const int pos = flat[hist[m]];
      const float* row = x + (size_t)pos * D_;
      if (j + 1 < 16) {
        const int posn = flat[hist[m + 1]];
        __builtin_prefetch(x + (size_t)posn * D_ + k0, 0, 1);   // global_prefetch_b8
      }
      float v[8];
      const float4 a0 = ((const float4*)(row + k0))[0];
      const float4 a1 = ((const float4*)(row + k0))[1];
      v[0] = a0.x; v[1] = a0.y; v[2] = a0.z; v[3] = a0.w;
      v[4] = a1.x; v[5] = a1.y; v[6] = a1.z; v[7] = a1.w;
      float s = 0.f, s2 = 0.f;
#pragma unroll
      for (int e = 0; e < 8; ++e) { s += v[e]; s2 += v[e] * v[e]; }
#pragma unroll
      for (int off = 16; off >= 1; off >>= 1) {
        s  += __shfl_xor(s,  off);
        s2 += __shfl_xor(s2, off);
      }
      const float mu  = s * (1.0f / D_);
      const float var = s2 * (1.0f / D_) - mu * mu;
      const float rs  = rsqrtf(var + 1e-5f);
#pragma unroll
      for (int e = 0; e < 8; ++e) {
        float xn = (v[e] - mu) * rs * lw[e] + lb[e];
        As[m * D_ + k0 + e] = (_Float16)xn;
      }
    }
  }

#if USE_TDM
  if (w == 0) __builtin_amdgcn_s_wait_tensorcnt(0);   // TDM done before barrier
#endif
  __syncthreads();

  // ---- WMMA GEMM: wave w owns M-tile w (16 tokens) x 8 N-tiles ----
  union Frag { v16h v; unsigned int u[8]; };
  v8f acc[8] = {};
  const int arow = (w * 16 + lane_r) * D_;
#pragma unroll
  for (int kt = 0; kt < 8; ++kt) {
    Frag a;
#pragma unroll
    for (int vi = 0; vi < 8; ++vi) {
      const int group = vi >> 2, vin = vi & 3;
      const int K0 = kt * 32 + lane_h * 16 + vi * 2;
      a.u[vi] = *(const unsigned int*)(&As[arow + K0]);
    }
#pragma unroll
    for (int nt = 0; nt < 8; ++nt) {
      Frag bf;
      const int nb = (nt * 16 + lane_r) * D_ + kt * 32 + lane_h * 16;
#pragma unroll
      for (int vi = 0; vi < 8; ++vi)
        bf.u[vi] = *(const unsigned int*)(&Bs[nb + vi * 2]);
      acc[nt] = __builtin_amdgcn_wmma_f32_16x16x32_f16(
          false, a.v, false, bf.v, (short)0, acc[nt], false, false);
    }
  }

  // ---- bias + exact GELU + column sums over this block's 128 tokens ----
  const float* b1 = (p ? eb1 : nb1) + l * H_;
#pragma unroll
  for (int nt = 0; nt < 8; ++nt) {
    const int n = nt * 16 + lane_r;
    const float bias = b1[n];
    float s = 0.f;
#pragma unroll
    for (int vi = 0; vi < 8; ++vi) {
      float h = acc[nt][vi] + bias;
      s += 0.5f * h * (1.0f + erff(h * 0.70710678118654752f));
    }
    s += __shfl_xor(s, 16);              // combine M=0..7 with M=8..15 halves
    if (lane_h == 0) atomicAdd(&hsum[n], s);
  }
  __syncthreads();
  if (tid < H_)
    atomicAdd(&hacc[((p * L_ + l) * T_ + t) * H_ + tid], hsum[tid]);
}

// ---------------------------------------------------------------------------
// Final: pooled = (hacc/1024) @ W2 + b2; write compressed; ortho loss.
// ---------------------------------------------------------------------------
__global__ __launch_bounds__(256)
void final_kernel(const float* __restrict__ hacc,
                  const float* __restrict__ nW2, const float* __restrict__ nb2,
                  const float* __restrict__ eW2, const float* __restrict__ eb2,
                  float* __restrict__ out) {
  __shared__ float yS[2 * L_ * T_ * C_];   // 8192 floats = 32 KB
  __shared__ float rn[2 * L_ * T_];
  __shared__ float osum;
  const int tid = threadIdx.x;
  if (tid == 0) osum = 0.f;

  for (int o = tid; o < 2 * L_ * T_ * C_; o += 256) {
    const int c = o & (C_ - 1);
    const int t = (o >> 6) & (T_ - 1);
    const int a = o >> 10;               // p*4 + l
    const int p = a >> 2, l = a & 3;
    const float* W2 = (p ? eW2 : nW2) + l * H_ * C_;
    const float* b2 = (p ? eb2 : nb2) + l * C_;
    const float* hb = hacc + (a * T_ + t) * H_;
    float s = b2[c];
    for (int k = 0; k < H_; ++k)
      s += (hb[k] * (1.0f / 1024.0f)) * W2[k * C_ + c];
    yS[o] = s;
    out[t * (2 * L_ * C_) + a * C_ + c] = s;   // compressed [T,1,512]
  }
  __syncthreads();

  if (tid < 2 * L_ * T_) {
    const float* y = yS + tid * C_;
    float s = 0.f;
    for (int c = 0; c < C_; ++c) s += y[c] * y[c];
    rn[tid] = 1.0f / fmaxf(sqrtf(s), 1e-12f);
  }
  __syncthreads();

  float part = 0.f;
  for (int e = tid; e < 7 * 7 * T_; e += 256) {   // G[:-1,1:] -> a:0..6, b:1..7
    const int t  = e & (T_ - 1);
    const int ab = e >> 4;
    const int a  = ab / 7;
    const int bb = 1 + (ab % 7);
    const float* ya = yS + (a * T_ + t) * C_;
    const float* yb = yS + (bb * T_ + t) * C_;
    float d = 0.f;
    for (int c = 0; c < C_; ++c) d += ya[c] * yb[c];
    d *= rn[a * T_ + t] * rn[bb * T_ + t];
    part += d * d;
  }
  atomicAdd(&osum, part);
  __syncthreads();
  if (tid == 0) out[2 * L_ * T_ * C_] = osum / (7.0f * 7.0f * (float)T_);
}

// ---------------------------------------------------------------------------
extern "C" void kernel_launch(void* const* d_in, const int* in_sizes, int n_in,
                              void* d_out, int out_size, void* d_ws, size_t ws_size,
                              hipStream_t stream) {
  const float* x    = (const float*)d_in[0];
  const float* lnw  = (const float*)d_in[1];
  const float* lnb  = (const float*)d_in[2];
  const float* nW1  = (const float*)d_in[3];
  const float* nb1  = (const float*)d_in[4];
  const float* nW2  = (const float*)d_in[5];
  const float* nb2  = (const float*)d_in[6];
  const float* eW1  = (const float*)d_in[7];
  const float* eb1  = (const float*)d_in[8];
  const float* eW2  = (const float*)d_in[9];
  const float* eb2  = (const float*)d_in[10];
  const int*   nflat = (const int*)d_in[11];
  const int*   eflat = (const int*)d_in[12];
  const int*   nhist = (const int*)d_in[13];
  const int*   ehist = (const int*)d_in[14];
  float* out = (float*)d_out;

  _Float16* w1h = (_Float16*)d_ws;
  float* hacc = (float*)((char*)d_ws + (size_t)W1H_ELEMS * sizeof(_Float16));

  const int prep_total = W1H_ELEMS + HACC_ELEMS;
  prep_kernel<<<(prep_total + 255) / 256, 256, 0, stream>>>(nW1, eW1, w1h, hacc);

  const int mlp_blocks = 2 * L_ * T_ * CHUNKS;   // 1024
  mlp_kernel<<<mlp_blocks, 256, 0, stream>>>(x, lnw, lnb, nb1, eb1,
                                             nflat, eflat, nhist, ehist,
                                             w1h, hacc);

  final_kernel<<<1, 256, 0, stream>>>(hacc, nW2, nb2, eW2, eb2, out);
}
